// RGCN_62775241998907
// MI455X (gfx1250) — matrix operand order repaired and checked by reading, not compiled, not run on hardware
//
#include <hip/hip_runtime.h>

typedef __attribute__((ext_vector_type(16))) __bf16 v16bf;
typedef __attribute__((ext_vector_type(8)))  float  v8f;

#define NNODES 100000
#define NEDGES 600000
#define NB 16
#define EPSBN 1e-5f

__device__ __forceinline__ __bf16 f2bf(float x) {
  unsigned int u = __builtin_bit_cast(unsigned int, x);
  unsigned int r = (u + 0x7FFFu + ((u >> 16) & 1u)) >> 16;
  return __builtin_bit_cast(__bf16, (unsigned short)r);
}
__device__ __forceinline__ unsigned int mapf(float f) {
  unsigned int u = __builtin_bit_cast(unsigned int, f);
  return (u & 0x80000000u) ? ~u : (u | 0x80000000u);
}
__device__ __forceinline__ float unmapf(unsigned int u) {
  unsigned int b = (u & 0x80000000u) ? (u ^ 0x80000000u) : ~u;
  return __builtin_bit_cast(float, b);
}

// Packed-B index: B[k][col] -> [ntile][kstep][lane][j] so each lane's 16 bf16 are
// contiguous (32B). lane = 16*((k>>4)&1) + (col&15), j = k&15 (contiguous k = 16*half + j, same as A loads).
__device__ __forceinline__ size_t bpack_idx(int k, int col, int Ksz) {
  int nt = col >> 4, l16 = col & 15, kbi = k >> 5, khalf = (k >> 4) & 1, j = k & 15;
  return ((size_t)(nt * (Ksz >> 5) + kbi) * 32 + khalf * 16 + l16) * 16 + j;
}

// ---------------- utility fills ----------------
__global__ void fill_u32_k(unsigned int* p, unsigned int v, long long n) {
  long long i = (long long)blockIdx.x * blockDim.x + threadIdx.x;
  if (i < n) p[i] = v;
}

// ---------------- weight prep ----------------
// W1[r][k][f] = sum_b comp1[r][b] * basis1[b][k][f]   (2 x 3 x 64, f32)
__global__ void prep_w1_k(const float* __restrict__ comp1, const float* __restrict__ basis1,
                          float* __restrict__ w1) {
  int t = blockIdx.x * blockDim.x + threadIdx.x;
  if (t >= 2 * 3 * 64) return;
  int f = t & 63, k = (t >> 6) % 3, r = t / 192;
  float s = 0.f;
  for (int b = 0; b < 64; ++b) s += comp1[r * 64 + b] * basis1[b * 192 + k * 64 + f];
  w1[t] = s;
}

// w2pack: packed bf16 of [W2_0 | W2_1 | root2]  (K=64, Nw=384)
__global__ void prep_w2cat_k(const float* __restrict__ comp2, const float* __restrict__ basis2,
                             const float* __restrict__ root2, __bf16* __restrict__ w2pack) {
  int t = blockIdx.x * blockDim.x + threadIdx.x;
  if (t >= 64 * 384) return;
  int col = t % 384, k = t / 384;
  float s;
  if (col < 256) {
    int r = col >> 7, o = col & 127;
    s = 0.f;
    for (int b = 0; b < 128; ++b) s += comp2[r * 128 + b] * basis2[b * 8192 + k * 128 + o];
  } else {
    s = root2[k * 128 + (col - 256)];
  }
  w2pack[bpack_idx(k, col, 64)] = f2bf(s);
}

// W_lin1 [128][1024] f32 -> packed bf16 (K=128, Nw=1024)
__global__ void pack_wlin1_k(const float* __restrict__ W, __bf16* __restrict__ wpack) {
  int t = blockIdx.x * blockDim.x + threadIdx.x;
  if (t >= 128 * 1024) return;
  int col = t & 1023, k = t >> 10;
  wpack[bpack_idx(k, col, 128)] = f2bf(W[t]);
}

// ---------------- RGCN layer 1 (K=3, plain VALU) ----------------
__global__ void layer1_node_k(const float* __restrict__ pos, const float* __restrict__ w1,
                              const float* __restrict__ root1, const float* __restrict__ bias1,
                              float* __restrict__ self1, float* __restrict__ t1) {
  int t = blockIdx.x * blockDim.x + threadIdx.x;
  if (t >= NNODES * 64) return;
  int f = t & 63, n = t >> 6;
  float p0 = pos[n * 3 + 0], p1 = pos[n * 3 + 1], p2 = pos[n * 3 + 2];
  self1[t] = bias1[f] + p0 * root1[f] + p1 * root1[64 + f] + p2 * root1[128 + f];
  for (int r = 0; r < 2; ++r)
    t1[(size_t)r * NNODES * 64 + t] =
        p0 * w1[r * 192 + f] + p1 * w1[r * 192 + 64 + f] + p2 * w1[r * 192 + 128 + f];
}

__global__ void layer1_edge_k(const int* __restrict__ ei, const int* __restrict__ et,
                              const float* __restrict__ t1, float* __restrict__ acc1) {
  int t = blockIdx.x * blockDim.x + threadIdx.x;
  if (t >= NEDGES * 4) return;
  int e = t >> 2, f0 = (t & 3) * 16;
  int src = ei[e], dst = ei[NEDGES + e], r = et[e];
  const float* s = t1 + (size_t)r * NNODES * 64 + (size_t)src * 64 + f0;
  float* a = acc1 + (size_t)dst * 128 + r * 64 + f0;
#pragma unroll
  for (int j = 0; j < 16; ++j) atomicAdd(&a[j], s[j]);
}

__global__ void edge_cnt_k(const int* __restrict__ ei, const int* __restrict__ et,
                           float* __restrict__ cnt) {
  int e = blockIdx.x * blockDim.x + threadIdx.x;
  if (e >= NEDGES) return;
  atomicAdd(&cnt[(size_t)et[e] * NNODES + ei[NEDGES + e]], 1.0f);
}

__global__ void layer1_final_k(const float* __restrict__ self1, const float* __restrict__ acc1,
                               const float* __restrict__ cnt, __bf16* __restrict__ x1bf) {
  int t = blockIdx.x * blockDim.x + threadIdx.x;
  if (t >= NNODES * 64) return;
  int f = t & 63, n = t >> 6;
  float i0 = 1.f / fmaxf(cnt[n], 1.f);
  float i1 = 1.f / fmaxf(cnt[NNODES + n], 1.f);
  float v = self1[t] + acc1[(size_t)n * 128 + f] * i0 + acc1[(size_t)n * 128 + 64 + f] * i1;
  x1bf[t] = f2bf(v);
}

// ---------------- bf16 WMMA GEMM: out[M x Nw] = A[M x K] * Bpack ----------------
// blockDim.x = 256 (8 waves). grid.x = M/16. Each wave: 16 x (16*NT) output strip.
// A loads: 2x b128 per K-step; B loads: 32B contiguous per lane from packed layout.
template <int K, int Nw, int NT>
__global__ void wmma_gemm_bf16_k(const __bf16* __restrict__ A, const __bf16* __restrict__ Bp,
                                 float* __restrict__ out, const float* __restrict__ bias,
                                 int biasStart, int doRelu,
                                 float* __restrict__ colsum, float* __restrict__ colsq) {
  constexpr int NTILES = Nw / 16;
  const int wave  = threadIdx.x >> 5;
  const int lane  = threadIdx.x & 31;
  const int l16   = lane & 15;
  const int khalf = lane >> 4;
  const int mtile = blockIdx.x;
  const int nt0   = (blockIdx.y * 8 + wave) * NT;
  if (nt0 >= NTILES) return;
  const int mrow = mtile * 16 + l16;

  v8f zero = {};
  v8f acc[NT];
#pragma unroll
  for (int q = 0; q < NT; ++q) acc[q] = zero;

#pragma unroll
  for (int kb = 0; kb < K; kb += 32) {
    v16bf av;
    const __bf16* ar = A + (size_t)mrow * K + kb + khalf * 16;
#pragma unroll
    for (int j = 0; j < 16; ++j) av[j] = ar[j];
#pragma unroll
    for (int q = 0; q < NT; ++q) {
      v16bf bv;
      const __bf16* bp = Bp + ((size_t)((nt0 + q) * (K >> 5) + (kb >> 5)) * 32 + lane) * 16;
#pragma unroll
      for (int j = 0; j < 16; ++j) bv[j] = bp[j];
      acc[q] = __builtin_amdgcn_wmma_f32_16x16x32_bf16(false, av, false, bv, (short)0,
                                                       acc[q], false, false);
    }
  }

#pragma unroll
  for (int q = 0; q < NT; ++q) {
    int col = (nt0 + q) * 16 + l16;
    float bsv = 0.f;
    if (bias != nullptr && col >= biasStart) bsv = bias[col - biasStart];
    float s = 0.f, s2 = 0.f;
#pragma unroll
    for (int i = 0; i < 8; ++i) {
      int row = mtile * 16 + khalf * 8 + i;
      float v = acc[q][i] + bsv;
      if (doRelu) v = fmaxf(v, 0.f);
      out[(size_t)row * Nw + col] = v;
      s += v;
      s2 += v * v;
    }
    if (colsum != nullptr) {  // fused BN column stats (16 rows per lane-pair)
      s  += __shfl_xor(s, 16, 32);
      s2 += __shfl_xor(s2, 16, 32);
      if (khalf == 0) {
        atomicAdd(&colsum[col], s);
        atomicAdd(&colsq[col], s2);
      }
    }
  }
}

// ---------------- RGCN layer 2 edge/final ----------------
__global__ void layer2_edge_k(const int* __restrict__ ei, const int* __restrict__ et,
                              const float* __restrict__ t2, float* __restrict__ acc2) {
  int t = blockIdx.x * blockDim.x + threadIdx.x;
  if (t >= NEDGES * 8) return;
  int e = t >> 3, f0 = (t & 7) * 16;
  int src = ei[e], dst = ei[NEDGES + e], r = et[e];
  const float* s = t2 + (size_t)src * 384 + r * 128 + f0;
  float* a = acc2 + (size_t)dst * 256 + r * 128 + f0;
#pragma unroll
  for (int j = 0; j < 16; ++j) atomicAdd(&a[j], s[j]);
}

__global__ void layer2_final_k(const float* __restrict__ t2, const float* __restrict__ acc2,
                               const float* __restrict__ cnt, __bf16* __restrict__ x2bf) {
  int t = blockIdx.x * blockDim.x + threadIdx.x;
  if (t >= NNODES * 128) return;
  int f = t & 127, n = t >> 7;
  float i0 = 1.f / fmaxf(cnt[n], 1.f);
  float i1 = 1.f / fmaxf(cnt[NNODES + n], 1.f);
  float v = t2[(size_t)n * 384 + 256 + f] + acc2[(size_t)n * 256 + f] * i0 +
            acc2[(size_t)n * 256 + 128 + f] * i1;
  x2bf[t] = f2bf(v);
}

// ---------------- normalize + segment_max pool ----------------
__global__ void pool_max_k(const float* __restrict__ h, const float* __restrict__ colsum,
                           const float* __restrict__ colsq, const float* __restrict__ g,
                           const float* __restrict__ be, const int* __restrict__ batch,
                           unsigned int* __restrict__ pmax) {
  long long t = (long long)blockIdx.x * blockDim.x + threadIdx.x;
  if (t >= (long long)NNODES * 1024) return;
  int c = (int)(t & 1023);
  int n = (int)(t >> 10);
  float m = colsum[c] * (1.f / NNODES);
  float var = colsq[c] * (1.f / NNODES) - m * m;
  float v = (h[t] - m) * rsqrtf(var + EPSBN) * g[c] + be[c];
  atomicMax(&pmax[(size_t)batch[n] * 1024 + c], mapf(v));
}

__global__ void unmap_pool_k(const unsigned int* __restrict__ pmax, float* __restrict__ pf) {
  int t = blockIdx.x * blockDim.x + threadIdx.x;
  if (t >= NB * 1024) return;
  float v = unmapf(pmax[t]);
  pf[t] = (v >= -3.4e38f) ? v : 0.f;  // -inf (empty graph) -> 0
}

// ---------------- tiny MLP head: Linear -> ReLU -> BN over 16 rows ----------------
template <int KDIM, int ODIM>
__global__ void head_mlp_k(const float* __restrict__ in, const float* __restrict__ W,
                           const float* __restrict__ bias, const float* __restrict__ g,
                           const float* __restrict__ be, float* __restrict__ outp) {
  int j = threadIdx.x;  // blockDim.x == ODIM
  if (j >= ODIM) return;
  float v[NB];
#pragma unroll 1
  for (int b = 0; b < NB; ++b) {
    float s = bias[j];
    for (int k = 0; k < KDIM; ++k) s += in[b * KDIM + k] * W[(size_t)k * ODIM + j];
    v[b] = fmaxf(s, 0.f);
  }
  float m = 0.f;
  for (int b = 0; b < NB; ++b) m += v[b];
  m *= (1.f / NB);
  float var = 0.f;
  for (int b = 0; b < NB; ++b) { float d = v[b] - m; var += d * d; }
  var *= (1.f / NB);
  float sc = rsqrtf(var + EPSBN) * g[j];
  for (int b = 0; b < NB; ++b) outp[b * ODIM + j] = (v[b] - m) * sc + be[j];
}

__global__ void head3_k(const float* __restrict__ h2, const float* __restrict__ W3,
                        const float* __restrict__ b3, float* __restrict__ out) {
  int t = threadIdx.x;
  if (t >= NB * 2) return;
  int b = t >> 1, c = t & 1;
  float s = b3[c];
  for (int k = 0; k < 256; ++k) s += h2[b * 256 + k] * W3[k * 2 + c];
  out[b * 2 + c] = s;
}

extern "C" void kernel_launch(void* const* d_in, const int* in_sizes, int n_in,
                              void* d_out, int out_size, void* d_ws, size_t ws_size,
                              hipStream_t stream) {
  const float* pos    = (const float*)d_in[0];
  const int*   ei     = (const int*)d_in[1];
  const int*   et     = (const int*)d_in[2];
  const int*   batch  = (const int*)d_in[3];
  const float* basis1 = (const float*)d_in[4];
  const float* comp1  = (const float*)d_in[5];
  const float* root1  = (const float*)d_in[6];
  const float* bias1  = (const float*)d_in[7];
  const float* basis2 = (const float*)d_in[8];
  const float* comp2  = (const float*)d_in[9];
  const float* root2  = (const float*)d_in[10];
  const float* bias2  = (const float*)d_in[11];
  const float* W_lin1 = (const float*)d_in[12];
  const float* b_lin1 = (const float*)d_in[13];
  const float* g_lin1 = (const float*)d_in[14];
  const float* be_lin1= (const float*)d_in[15];
  const float* W1h    = (const float*)d_in[16];
  const float* b1h    = (const float*)d_in[17];
  const float* g1h    = (const float*)d_in[18];
  const float* be1h   = (const float*)d_in[19];
  const float* W2h    = (const float*)d_in[20];
  const float* b2h    = (const float*)d_in[21];
  const float* g2h    = (const float*)d_in[22];
  const float* be2h   = (const float*)d_in[23];
  const float* W3h    = (const float*)d_in[24];
  const float* b3h    = (const float*)d_in[25];
  float* out = (float*)d_out;

  // ---- workspace carve-out (256B aligned) ----
  char* ws = (char*)d_ws;
  size_t off = 0;
  auto alloc = [&](size_t bytes) -> void* {
    void* p = ws + off;
    off = (off + bytes + 255) & ~(size_t)255;
    return p;
  };

  // Union region: phase-1 holds acc1|cnt|acc2|t1|t2 (89.8M f32); phase-2 holds h (102.4M f32).
  float* regionA = (float*)alloc(102400000ull * 4);
  float* acc1 = regionA;                 // N*128
  float* cnt  = regionA + 12800000;      // 2*N
  float* acc2 = regionA + 13000000;      // N*256
  float* t1   = regionA + 38600000;      // 2*N*64
  float* t2   = regionA + 51400000;      // N*384
  float* h    = regionA;                 // N*1024 (phase 2)

  float*  w1      = (float*)alloc(384 * 4);
  __bf16* w2pack  = (__bf16*)alloc(24576 * 2);
  __bf16* wlpack  = (__bf16*)alloc(131072 * 2);
  float*  self1   = (float*)alloc(6400000ull * 4);
  __bf16* x1bf    = (__bf16*)alloc(6400000ull * 2);
  __bf16* x2bf    = (__bf16*)alloc(12800000ull * 2);
  float*  colsum  = (float*)alloc(1024 * 4);
  float*  colsq   = (float*)alloc(1024 * 4);
  unsigned int* pmax = (unsigned int*)alloc(16384 * 4);
  float*  pf      = (float*)alloc(16384 * 4);
  float*  h1      = (float*)alloc(8192 * 4);
  float*  h2      = (float*)alloc(4096 * 4);

  // ---- zero / init ----
  fill_u32_k<<<(38600000 + 255) / 256, 256, 0, stream>>>((unsigned int*)acc1, 0u, 38600000ll);
  fill_u32_k<<<4, 256, 0, stream>>>((unsigned int*)colsum, 0u, 1024ll);
  fill_u32_k<<<4, 256, 0, stream>>>((unsigned int*)colsq, 0u, 1024ll);
  fill_u32_k<<<64, 256, 0, stream>>>(pmax, 0x007FFFFFu, 16384ll);  // mapped(-inf)

  // ---- weight prep (packed-B layouts for WMMA) ----
  prep_w1_k<<<2, 256, 0, stream>>>(comp1, basis1, w1);
  prep_w2cat_k<<<96, 256, 0, stream>>>(comp2, basis2, root2, w2pack);
  pack_wlin1_k<<<512, 256, 0, stream>>>(W_lin1, wlpack);

  // ---- RGCN layer 1 ----
  layer1_node_k<<<25000, 256, 0, stream>>>(pos, w1, root1, bias1, self1, t1);
  layer1_edge_k<<<(NEDGES * 4 + 255) / 256, 256, 0, stream>>>(ei, et, t1, acc1);
  edge_cnt_k<<<(NEDGES + 255) / 256, 256, 0, stream>>>(ei, et, cnt);
  layer1_final_k<<<25000, 256, 0, stream>>>(self1, acc1, cnt, x1bf);

  // ---- RGCN layer 2: t2[N,384] = x1 @ [W2_0|W2_1|root2] (+bias2 on cols>=256) ----
  wmma_gemm_bf16_k<64, 384, 4><<<dim3(6250, 1), 256, 0, stream>>>(
      x1bf, w2pack, t2, bias2, 256, 0, nullptr, nullptr);
  layer2_edge_k<<<(NEDGES * 8 + 255) / 256, 256, 0, stream>>>(ei, et, t2, acc2);
  layer2_final_k<<<50000, 256, 0, stream>>>(t2, acc2, cnt, x2bf);

  // ---- lin1: h = relu(x2 @ W_lin1 + b), with fused BN column stats ----
  wmma_gemm_bf16_k<128, 1024, 4><<<dim3(6250, 2), 256, 0, stream>>>(
      x2bf, wlpack, h, b_lin1, 0, 1, colsum, colsq);

  // ---- normalize + global max pool ----
  pool_max_k<<<400000, 256, 0, stream>>>(h, colsum, colsq, g_lin1, be_lin1, batch, pmax);
  unmap_pool_k<<<64, 256, 0, stream>>>(pmax, pf);

  // ---- head MLP ----
  head_mlp_k<1024, 512><<<1, 512, 0, stream>>>(pf, W1h, b1h, g1h, be1h, h1);
  head_mlp_k<512, 256><<<1, 256, 0, stream>>>(h1, W2h, b2h, g2h, be2h, h2);
  head3_k<<<1, 32, 0, stream>>>(h2, W3h, b3h, out);
}
